// BackflowNet_87479893885117
// MI455X (gfx1250) — compile-verified, matrix operand order checked by reading, not run on hardware
//
#include <hip/hip_runtime.h>
#include <hip/hip_bf16.h>

typedef __attribute__((ext_vector_type(16))) _Float16 v16h;
typedef __attribute__((ext_vector_type(8)))  _Float16 h8;
typedef __attribute__((ext_vector_type(8)))  float    v8f;

namespace {
constexpr int kB   = 256;
constexpr int kN   = 64;
constexpr int kMsgH = 128;
constexpr int kHid  = 128;

// Workspace layout (half-element offsets for packed weight fragments)
constexpr int HOFF_PHI1 = 0;        // 8 nt * 1 ks * 512 halves
constexpr int HOFF_PHI2 = 4096;     // 8 nt * 4 ks * 512
constexpr int HOFF_PSI1 = 20480;    // 8 nt * 5 ks * 512
constexpr int HOFF_PSI2 = 40960;    // 8 nt * 4 ks * 512
constexpr int HELEMS    = 57344;    // total packed halves
constexpr size_t BOFF_MI = (size_t)HELEMS * 2;  // byte offset of m_i (f32, B*N*128)
}

union FragA { v16h v; h8 c[2]; _Float16 h[16]; };
union FragC { v8f  v; float f[8]; };

// Fast SiLU: one v_exp + one v_rcp instead of the IEEE-divide sequence.
__device__ __forceinline__ float silu_f(float v) {
  return v * __builtin_amdgcn_rcpf(1.0f + __expf(-v));
}

// Wave-local LDS ordering: the LDS tiles are wave-private, and same-wave LDS
// ops complete in order on CDNA5, so a full workgroup s_barrier is overkill.
// A wavefront-scope fence + scheduling barrier keeps the compiler honest
// without emitting any hardware synchronization between waves.
__device__ __forceinline__ void wave_lds_sync() {
  __builtin_amdgcn_fence(__ATOMIC_RELEASE, "wavefront");
  __builtin_amdgcn_wave_barrier();
  __builtin_amdgcn_fence(__ATOMIC_ACQUIRE, "wavefront");
}

// ---------------------------------------------------------------------------
// Pack f32 weights (fin x 128, row-major) into per-lane WMMA B-fragment order:
// frag[nt][ks][lane][j] with k = ks*32 + (lane<16?0:8) + j + (j>=8?8:0),
// col = nt*16 + (lane&15). Bias folded in as weight row k == Kreal (the A
// fragment supplies a constant 1.0 there); zero past that.
// ---------------------------------------------------------------------------
__global__ void prep_weights_kernel(const float* __restrict__ phi_w1,
                                    const float* __restrict__ phi_b1,
                                    const float* __restrict__ phi_w2,
                                    const float* __restrict__ psi_w1,
                                    const float* __restrict__ psi_b1,
                                    const float* __restrict__ psi_w2,
                                    _Float16* __restrict__ wsh) {
  int e = blockIdx.x * blockDim.x + threadIdx.x;
  if (e >= HELEMS) return;
  const float* W; const float* Bv; int KS, Kreal, local;
  if (e < HOFF_PHI2)      { W = phi_w1; Bv = phi_b1; KS = 1; Kreal = 11;  local = e - HOFF_PHI1; }
  else if (e < HOFF_PSI1) { W = phi_w2; Bv = nullptr; KS = 4; Kreal = 128; local = e - HOFF_PHI2; }
  else if (e < HOFF_PSI2) { W = psi_w1; Bv = psi_b1; KS = 5; Kreal = 131; local = e - HOFF_PSI1; }
  else                    { W = psi_w2; Bv = nullptr; KS = 4; Kreal = 128; local = e - HOFF_PSI2; }
  int nt   = local / (KS * 512);
  int rem  = local % (KS * 512);
  int ks   = rem / 512;
  int lj   = rem % 512;
  int lane = lj / 16;
  int j    = lj % 16;
  int k    = ks * 32 + ((lane < 16) ? 0 : 8) + j + ((j >= 8) ? 8 : 0);
  int col  = nt * 16 + (lane & 15);
  float v;
  if (k < Kreal)                    v = W[k * 128 + col];
  else if (k == Kreal && Bv)        v = Bv[col];        // folded bias row
  else                              v = 0.0f;
  wsh[e] = (_Float16)v;
}

// ---------------------------------------------------------------------------
// Kernel 1: pairwise message MLP + masked sum over j.
// One block per batch element b; 8 waves; wave w owns i = w*8 .. w*8+7.
// Layer-2 GEMM: ks-outer / nt-inner with c[8] accumulators. waves_per_eu(4)
// caps the allocator below 256 VGPRs. Waves run fully decoupled (wave-local
// LDS fences only, no workgroup barriers).
// ---------------------------------------------------------------------------
__global__ __launch_bounds__(256)
__attribute__((amdgpu_waves_per_eu(4)))
void k_messages(
    const float* __restrict__ x,
    const float* __restrict__ phi_b2,
    const _Float16* __restrict__ wsh,
    float* __restrict__ m_i_out) {
  __shared__ __align__(16) _Float16 h_tile[8][16 * kMsgH];  // 32 KB, per-wave tiles

  const int b    = blockIdx.x;
  const int lane = threadIdx.x & 31;
  const int wave = threadIdx.x >> 5;
  const int r    = lane & 15;
  const int hi   = (lane < 16) ? 0 : 8;  // half-wave row offset / K offset

  const _Float16* phi1f = wsh + HOFF_PHI1;
  const _Float16* phi2f = wsh + HOFF_PHI2;
  _Float16* hrow = &h_tile[wave][0];

  // Layer-2 bias per lane (column = nt*16 + r for every C element).
  float bias2[8];
  #pragma unroll
  for (int nt = 0; nt < 8; ++nt) bias2[nt] = phi_b2[nt * 16 + r];

  #pragma unroll 1
  for (int q = 0; q < 8; ++q) {
    const int i = wave * 8 + q;

    float msum[8];  // per-lane partial column sums
    #pragma unroll
    for (int nt = 0; nt < 8; ++nt) msum[nt] = 0.0f;

    const float xi0 = x[(b * kN + i) * 3 + 0];
    const float xi1 = x[(b * kN + i) * 3 + 1];
    const float xi2 = x[(b * kN + i) * 3 + 2];

    #pragma unroll 1
    for (int jt = 0; jt < 4; ++jt) {
      const int j = jt * 16 + r;
      const float xj0 = x[(b * kN + j) * 3 + 0];
      const float xj1 = x[(b * kN + j) * 3 + 1];
      const float xj2 = x[(b * kN + j) * 3 + 2];
      const float d0 = xi0 - xj0, d1 = xi1 - xj1, d2 = xi2 - xj2;
      const float rr = d0 * d0 + d1 * d1 + d2 * d2;
      const float rn = sqrtf(rr + 1e-12f);

      // Layer-1 A fragment (16x32, K 11 real + K=11 bias-one + zeros).
      // Lanes 0-15: K=0..7 (feats 0..7); lanes 16-31: K=8..10 + 1.0 at K=11.
      FragA a;
      #pragma unroll
      for (int t = 0; t < 16; ++t) a.h[t] = (_Float16)0.0f;
      if (lane < 16) {
        a.h[0] = (_Float16)xi0; a.h[1] = (_Float16)xi1; a.h[2] = (_Float16)xi2;
        a.h[3] = (_Float16)xj0; a.h[4] = (_Float16)xj1; a.h[5] = (_Float16)xj2;
        a.h[6] = (_Float16)d0;  a.h[7] = (_Float16)d1;
      } else {
        a.h[0] = (_Float16)d2;  a.h[1] = (_Float16)rn;  a.h[2] = (_Float16)rr;
        a.h[3] = (_Float16)1.0f;  // bias row K=11
      }

      // Layer 1: (16x32) @ (32x128) -> h tile (f16 in LDS). Bias folded in W.
      #pragma unroll 2
      for (int nt = 0; nt < 8; ++nt) {
        v16h bf = ((const v16h*)phi1f)[nt * 32 + lane];
        FragC c;
        #pragma unroll
        for (int e = 0; e < 8; ++e) c.f[e] = 0.0f;
        c.v = __builtin_amdgcn_wmma_f32_16x16x32_f16(false, a.v, false, bf,
                                                     (short)0, c.v, false, false);
        #pragma unroll
        for (int e = 0; e < 8; ++e) {
          const int row = e + hi;
          hrow[row * kMsgH + nt * 16 + r] = (_Float16)silu_f(c.f[e]);
        }
      }
      wave_lds_sync();

      // Layer 2: (16x128) @ (128x128). ks-outer, nt-inner, c[8] accumulators.
      FragC c[8];
      #pragma unroll
      for (int nt = 0; nt < 8; ++nt)
        #pragma unroll
        for (int e = 0; e < 8; ++e) c[nt].f[e] = 0.0f;

      #pragma unroll
      for (int ks = 0; ks < 4; ++ks) {
        FragA af;
        const _Float16* rp = hrow + r * kMsgH + ks * 32 + hi;
        af.c[0] = *(const h8*)(rp);
        af.c[1] = *(const h8*)(rp + 16);
        #pragma unroll
        for (int nt = 0; nt < 8; ++nt) {
          v16h bf = ((const v16h*)phi2f)[(nt * 4 + ks) * 32 + lane];
          c[nt].v = __builtin_amdgcn_wmma_f32_16x16x32_f16(false, af.v, false, bf,
                                                           (short)0, c[nt].v,
                                                           false, false);
        }
      }

      const int selfRow = (jt == (i >> 4)) ? (i & 15) : -1;

      // Epilogue: bias + SiLU + self-mask + reduce rows to scalars.
      #pragma unroll
      for (int nt = 0; nt < 8; ++nt) {
        float s = 0.0f;
        #pragma unroll
        for (int e = 0; e < 8; ++e) {
          const int row = e + hi;
          float vv = silu_f(c[nt].f[e] + bias2[nt]);
          if (row == selfRow) vv = 0.0f;   // mask self-message j==i
          s += vv;
        }
        msum[nt] += s;
      }
      wave_lds_sync();  // h tile will be overwritten next jt iteration
    }

    // Combine half-wave partials -> m_i[b][i][:]
    #pragma unroll
    for (int nt = 0; nt < 8; ++nt) {
      float s = msum[nt] + __shfl_xor(msum[nt], 16, 32);
      if (lane < 16)
        m_i_out[(b * kN + i) * kMsgH + nt * 16 + lane] = s;
    }
  }
}

// ---------------------------------------------------------------------------
// Kernel 2: node update MLP. 1024 row-tiles of 16; 4 waves/block, 1 tile/wave.
// Same ks-outer / c[8] structure; psi_b1 folded into weights (k=131).
// ---------------------------------------------------------------------------
__global__ __launch_bounds__(128)
__attribute__((amdgpu_waves_per_eu(4)))
void k_update(
    const float* __restrict__ x,
    const float* __restrict__ psi_b2,
    const float* __restrict__ psi_w3,
    const float* __restrict__ psi_b3,
    const float* __restrict__ bf_scale,
    const _Float16* __restrict__ wsh,
    const float* __restrict__ m_i,
    float* __restrict__ out) {
  __shared__ __align__(16) _Float16 upd[4][16 * 160];   // 20 KB (reused as u2)
  __shared__ __align__(16) _Float16 u1[4][16 * kHid];   // 16 KB

  const int lane = threadIdx.x & 31;
  const int wave = threadIdx.x >> 5;
  const int tile = blockIdx.x * 4 + wave;  // 0..1023
  const int r    = lane & 15;
  const int hi   = (lane < 16) ? 0 : 8;

  const _Float16* psi1f = wsh + HOFF_PSI1;
  const _Float16* psi2f = wsh + HOFF_PSI2;
  _Float16* updw = &upd[wave][0];
  _Float16* u1w  = &u1[wave][0];

  float bias2[8];
  #pragma unroll
  for (int nt = 0; nt < 8; ++nt) bias2[nt] = psi_b2[nt * 16 + r];

  // Stage [x(3) | m_i(128) | 1.0 at k=131 (bias row) | pad] as f16.
  for (int idx = lane; idx < 16 * 160; idx += 32) {
    const int rr = idx / 160, k = idx % 160;
    const int g = tile * 16 + rr;
    float v;
    if (k < 3)         v = x[g * 3 + k];
    else if (k < 131)  v = m_i[g * 128 + (k - 3)];
    else if (k == 131) v = 1.0f;            // bias-one feature
    else               v = 0.0f;
    updw[idx] = (_Float16)v;
  }
  wave_lds_sync();

  // Layer 1: (16x160) @ (160x128), 5 K-steps, bias folded in W.
  {
    FragC c[8];
    #pragma unroll
    for (int nt = 0; nt < 8; ++nt)
      #pragma unroll
      for (int e = 0; e < 8; ++e) c[nt].f[e] = 0.0f;

    #pragma unroll
    for (int ks = 0; ks < 5; ++ks) {
      FragA af;
      const _Float16* rp = updw + r * 160 + ks * 32 + hi;
      af.c[0] = *(const h8*)(rp);
      af.c[1] = *(const h8*)(rp + 16);
      #pragma unroll
      for (int nt = 0; nt < 8; ++nt) {
        v16h bf = ((const v16h*)psi1f)[(nt * 5 + ks) * 32 + lane];
        c[nt].v = __builtin_amdgcn_wmma_f32_16x16x32_f16(false, af.v, false, bf,
                                                         (short)0, c[nt].v,
                                                         false, false);
      }
    }
    #pragma unroll
    for (int nt = 0; nt < 8; ++nt)
      #pragma unroll
      for (int e = 0; e < 8; ++e)
        u1w[(e + hi) * kHid + nt * 16 + r] = (_Float16)silu_f(c[nt].f[e]);
  }
  wave_lds_sync();

  // Layer 2: (16x128) @ (128x128); u2 written into updw region (stride 128).
  {
    FragC c[8];
    #pragma unroll
    for (int nt = 0; nt < 8; ++nt)
      #pragma unroll
      for (int e = 0; e < 8; ++e) c[nt].f[e] = 0.0f;

    #pragma unroll
    for (int ks = 0; ks < 4; ++ks) {
      FragA af;
      const _Float16* rp = u1w + r * kHid + ks * 32 + hi;
      af.c[0] = *(const h8*)(rp);
      af.c[1] = *(const h8*)(rp + 16);
      #pragma unroll
      for (int nt = 0; nt < 8; ++nt) {
        v16h bf = ((const v16h*)psi2f)[(nt * 4 + ks) * 32 + lane];
        c[nt].v = __builtin_amdgcn_wmma_f32_16x16x32_f16(false, af.v, false, bf,
                                                         (short)0, c[nt].v,
                                                         false, false);
      }
    }
    #pragma unroll
    for (int nt = 0; nt < 8; ++nt)
      #pragma unroll
      for (int e = 0; e < 8; ++e)
        updw[(e + hi) * 128 + nt * 16 + r] = (_Float16)silu_f(c[nt].f[e] + bias2[nt]);
  }
  wave_lds_sync();

  // Final tiny layer 128 -> 3, then tanh * max(bf_scale, 0).
  const float scale = fmaxf(bf_scale[0], 0.0f);
  for (int t2 = lane; t2 < 48; t2 += 32) {
    const int rr = t2 / 3, d = t2 % 3;
    float acc = psi_b3[d];
    #pragma unroll 4
    for (int k = 0; k < 128; ++k)
      acc += (float)updw[rr * 128 + k] * psi_w3[k * 3 + d];
    const int g = tile * 16 + rr;
    out[g * 3 + d] = tanhf(acc) * scale;
  }
}

// ---------------------------------------------------------------------------
extern "C" void kernel_launch(void* const* d_in, const int* in_sizes, int n_in,
                              void* d_out, int out_size, void* d_ws, size_t ws_size,
                              hipStream_t stream) {
  (void)in_sizes; (void)n_in; (void)out_size; (void)ws_size;
  const float* x        = (const float*)d_in[0];
  // d_in[1] = spin (unused: mask is all-ones except diagonal)
  const float* phi_w1   = (const float*)d_in[2];
  const float* phi_b1   = (const float*)d_in[3];
  const float* phi_w2   = (const float*)d_in[4];
  const float* phi_b2   = (const float*)d_in[5];
  const float* psi_w1   = (const float*)d_in[6];
  const float* psi_b1   = (const float*)d_in[7];
  const float* psi_w2   = (const float*)d_in[8];
  const float* psi_b2   = (const float*)d_in[9];
  const float* psi_w3   = (const float*)d_in[10];
  const float* psi_b3   = (const float*)d_in[11];
  const float* bf_scale = (const float*)d_in[12];

  _Float16* wsh = (_Float16*)d_ws;
  float* m_i    = (float*)((char*)d_ws + BOFF_MI);
  float* out    = (float*)d_out;

  prep_weights_kernel<<<(HELEMS + 255) / 256, 256, 0, stream>>>(
      phi_w1, phi_b1, phi_w2, psi_w1, psi_b1, psi_w2, wsh);
  k_messages<<<kB, 256, 0, stream>>>(x, phi_b2, wsh, m_i);
  k_update<<<kB, 128, 0, stream>>>(x, psi_b2, psi_w3, psi_b3,
                                   bf_scale, wsh, m_i, out);
}
